// DSSKernel_704374636638
// MI455X (gfx1250) — compile-verified, matrix operand order checked by reading, not run on hardware
//
#include <hip/hip_runtime.h>

// ---------------------------------------------------------------------------
// DSS kernel for MI455X (gfx1250, wave32).
//   out[l,h] = Re( sum_n Wk[h,n] * z[h,n]^l ),  z = exp(dt_Lambda)
// Strategy: complex-multiply recurrence per (h,n); 64-way n-reduction done as
// P(16x32) x ones(32x16) via v_wmma_f32_16x16x32_f16 (f32 accumulate), with
// the lane->K transpose staged through a per-wave LDS buffer laid out so each
// lane's WMMA A-fragment is one contiguous 32B LDS load.
// ---------------------------------------------------------------------------

typedef __attribute__((ext_vector_type(16))) _Float16 v16h;
typedef __attribute__((ext_vector_type(8)))  float    v8f;

#define H_DIM   1024
#define N_DIM   64
#define L_DIM   2048
#define WAVES   16           // waves per workgroup (one h per wave)
#define LCHUNK  16           // l-steps per WMMA reduction
#define LT      (WAVES*LCHUNK) // 256 l values per workgroup

__global__ void __launch_bounds__(32*WAVES)
dss_kernel(const float* __restrict__ log_dt,          // (H,2)
           const float* __restrict__ lam_log_neg_re,  // (N,)
           const float* __restrict__ lam_im_in,       // (N,)
           const float* __restrict__ Wm,              // (H,N,2)
           float* __restrict__ out)                   // (L,H)
{
    // Per-wave transpose buffer: 16 rows (r) x 32 halves, permuted within row
    // so that lane L<16 reads halves [0..15] of row L  (= K 0..7,16..23) and
    // lane L>=16 reads halves [16..31] of row L-16 (= K 8..15,24..31): this is
    // exactly the 16-bit 16x32 WMMA A-matrix register layout.
    __shared__ _Float16 tbuf[WAVES][16 * 32];     // 16 KB
    __shared__ float    stile[16][17];            // [r][hh] output staging

    const int tid   = threadIdx.x;
    const int wave  = tid >> 5;
    const int lane  = tid & 31;
    const int h     = blockIdx.x * WAVES + wave;
    const int lbase = blockIdx.y * LT;

    // ---- per-h step sizes (sep_dt_re_im) ----
    const float dt0 = __expf(log_dt[2 * h + 0]);
    const float dt1 = __expf(log_dt[2 * h + 1]);

    // ---- per-lane state for its two n values (n = lane, lane+32) ----
    float zr[2], zi[2], cr[2], ci[2];
#pragma unroll
    for (int s = 0; s < 2; ++s) {
        const int n   = lane + 32 * s;
        const float lre = -__expf(lam_log_neg_re[n]);
        const float lim = lam_im_in[n];
        const float a = dt0 * lre;          // Re(dt_Lambda)
        const float b = dt1 * lim;          // Im(dt_Lambda)
        float sb, cb;
        const float ea = __expf(a);
        __sincosf(b, &sb, &cb);
        const float z_r = ea * cb, z_i = ea * sb;     // z = exp(dt_Lambda)
        // Wk = Wc * (z - 1) * conj(Lam) / max(|Lam|^2, eps^2)
        const float wr = Wm[(h * N_DIM + n) * 2 + 0];
        const float wi = Wm[(h * N_DIM + n) * 2 + 1];
        const float nsq = fmaxf(lre * lre + lim * lim, 1e-14f); // eps=1e-7
        const float rr =  lre / nsq;
        const float ri = -lim / nsq;
        const float er = z_r - 1.0f, ei = z_i;
        const float tr = wr * er - wi * ei;
        const float ti = wr * ei + wi * er;
        const float kr = tr * rr - ti * ri;
        const float ki = tr * ri + ti * rr;
        // seed state c = Wk * z^lbase  (direct exp/sincos for the jump)
        const float lf = (float)lbase;
        float s0, c0;
        const float e0 = __expf(a * lf);
        __sincosf(b * lf, &s0, &c0);
        const float pr = e0 * c0, pi = e0 * s0;
        cr[s] = kr * pr - ki * pi;
        ci[s] = kr * pi + ki * pr;
        zr[s] = z_r;  zi[s] = z_i;
    }

    // within-row LDS position for this lane's K column (see layout note above)
    const int tpos  = (lane >= 8 && lane < 24) ? (lane ^ 24) : lane;
    const int arow  = lane & 15;
    const int abase = (lane < 16) ? 0 : 16;
    const v16h* ap  = (const v16h*)&tbuf[wave][arow * 32 + abase];

    v16h ones;
#pragma unroll
    for (int i = 0; i < 16; ++i) ones[i] = (_Float16)1.0f;

    for (int chunk = 0; chunk < WAVES; ++chunk) {
        // ---- 16 recurrence steps: emit f16 partials into transpose buffer ----
#pragma unroll
        for (int r = 0; r < LCHUNK; ++r) {
            const float p = cr[0] + cr[1];            // Re contribution of 2 n's
            tbuf[wave][r * 32 + tpos] = (_Float16)p;
#pragma unroll
            for (int s = 0; s < 2; ++s) {             // c <- c * z
                const float nr = cr[s] * zr[s] - ci[s] * zi[s];
                const float ni = cr[s] * zi[s] + ci[s] * zr[s];
                cr[s] = nr;  ci[s] = ni;
            }
        }

        // same-wave DS ordering is in-order; fence the compiler + wait counter
        asm volatile("s_wait_dscnt 0" ::: "memory");

        // ---- WMMA reduction: D[r, *] = sum_k A[r,k] * 1 ----
        v16h a = *ap;
        v8f acc = {};
        acc = __builtin_amdgcn_wmma_f32_16x16x32_f16(
                  false, a, false, ones, (short)0, acc, false, false);

        // D layout: VGPR j = row (j | lane>=16 ? +8), value replicated over N.
        if (lane == 0) {
#pragma unroll
            for (int j = 0; j < 8; ++j) stile[j][wave] = acc[j];
        } else if (lane == 16) {
#pragma unroll
            for (int j = 0; j < 8; ++j) stile[8 + j][wave] = acc[j];
        }
        __syncthreads();

        // ---- coalesced store: adjacent lanes -> adjacent h ----
        if (tid < 256) {
            const int r  = tid >> 4;
            const int hh = tid & 15;
            const int l  = lbase + chunk * LCHUNK + r;
            out[l * H_DIM + blockIdx.x * WAVES + hh] = stile[r][hh];
        }
        __syncthreads();
    }
}

extern "C" void kernel_launch(void* const* d_in, const int* in_sizes, int n_in,
                              void* d_out, int out_size, void* d_ws, size_t ws_size,
                              hipStream_t stream) {
    (void)in_sizes; (void)n_in; (void)out_size; (void)d_ws; (void)ws_size;
    const float* log_dt = (const float*)d_in[0];   // (H,2)
    const float* lnr    = (const float*)d_in[1];   // (N,)
    const float* lim    = (const float*)d_in[2];   // (N,)
    const float* Wm     = (const float*)d_in[3];   // (H,N,2)
    float* out = (float*)d_out;                    // (L,H) fp32

    dim3 grid(H_DIM / WAVES, L_DIM / LT);          // 64 x 8 workgroups
    dss_kernel<<<grid, 32 * WAVES, 0, stream>>>(log_dt, lnr, lim, Wm, out);
}
